// CoAtNetCore_79688823210497
// MI455X (gfx1250) — matrix operand order repaired and checked by reading, not compiled, not run on hardware
//
#include <hip/hip_runtime.h>
#include <hip/hip_bf16.h>
#include <math.h>

typedef __attribute__((ext_vector_type(16))) _Float16 v16h;
typedef __attribute__((ext_vector_type(8)))  _Float16 v8h;
typedef __attribute__((ext_vector_type(8)))  float    v8f;
typedef int v4i __attribute__((vector_size(16)));
typedef __attribute__((address_space(1))) v4i* as1_v4i;
typedef __attribute__((address_space(3))) v4i* as3_v4i;

#define B_SZ  64
#define C_SZ  384
#define NTOK  256
#define HEADS 8
#define DH    32
#define INNER 256
#define HID   1536
#define MTOT  (B_SZ * NTOK)
#define QKV_W (3 * INNER)   // 768

#if __has_builtin(__builtin_amdgcn_global_load_async_to_lds_b128)
#define HAVE_ASYNC 1
#else
#define HAVE_ASYNC 0
#endif

// ---------------------------------------------------------------------------
// Async-counter waits (guarded; fall back to inline SOPP asm, then no-op)
// ---------------------------------------------------------------------------
static __device__ __forceinline__ void wait_async_le1() {
#if HAVE_ASYNC
#if __has_builtin(__builtin_amdgcn_s_wait_asynccnt)
  __builtin_amdgcn_s_wait_asynccnt(1);
#else
  asm volatile("s_wait_asynccnt 0x1" ::: "memory");
#endif
#endif
}
static __device__ __forceinline__ void wait_async_le0() {
#if HAVE_ASYNC
#if __has_builtin(__builtin_amdgcn_s_wait_asynccnt)
  __builtin_amdgcn_s_wait_asynccnt(0);
#else
  asm volatile("s_wait_asynccnt 0x0" ::: "memory");
#endif
#endif
}

// ---------------------------------------------------------------------------
// WMMA fragment loaders (CDNA5 16x16x32 f16 layouts, wave32)
// A: M x K row-major. lane holds row (lane&15); halves [0..7]=K k0+8*(lane>>4)..,
//    halves [8..15]=same+16.
// B: K x N, supplied as BT[N][K] row-major (i.e. column-major B). lane holds
//    col (lane&15); halves [0..7]=K k0+8*(lane>>4).., halves [8..15]=same+16.
// ---------------------------------------------------------------------------
static __device__ __forceinline__ v16h load_a_frag(const _Float16* A, int lda,
                                                   int row0, int k0, int lane) {
  int r  = row0 + (lane & 15);
  int kb = k0 + ((lane >> 4) << 3);
  const _Float16* p = A + (size_t)r * lda + kb;
  v8h lo = *(const v8h*)p;
  v8h hi = *(const v8h*)(p + 16);
  v16h o;
#pragma unroll
  for (int i = 0; i < 8; ++i) { o[i] = lo[i]; o[i + 8] = hi[i]; }
  return o;
}

static __device__ __forceinline__ v16h load_b_frag(const _Float16* BT, int ldb,
                                                   int col0, int k0, int lane) {
  int c  = col0 + (lane & 15);
  int kb = k0 + ((lane >> 4) << 3);
  const _Float16* p = BT + (size_t)c * ldb + kb;
  v8h lo = *(const v8h*)p;
  v8h hi = *(const v8h*)(p + 16);
  v16h o;
#pragma unroll
  for (int i = 0; i < 8; ++i) { o[i] = lo[i]; o[i + 8] = hi[i]; }
  return o;
}

// ---------------------------------------------------------------------------
// Stage one 64-col x 32-k B tile (f16, [col][k] layout, 4 KB) into LDS.
// 256 threads: thread t -> col t>>2, k-chunk (t&3)*8 -> one 16-byte transfer.
// Uses GLOBAL_LOAD_ASYNC_TO_LDS_B128 when available (ASYNCcnt-tracked DMA).
// AS1 pointer = flat address (identity for global); AS3 pointer = low 32 bits
// of the flat LDS address.
// ---------------------------------------------------------------------------
static __device__ __forceinline__ void stage_b(const _Float16* __restrict__ BT,
                                               _Float16* buf, int K, int col0,
                                               int k0, int tid) {
  int col = tid >> 2;
  int kc  = (tid & 3) << 3;
  const _Float16* src = BT + (size_t)(col0 + col) * K + k0 + kc;
  _Float16* dst = buf + col * 32 + kc;
#if HAVE_ASYNC
  as1_v4i gsrc = (as1_v4i)(unsigned long long)src;
  as3_v4i ldst = (as3_v4i)(unsigned int)(unsigned long long)dst;
  __builtin_amdgcn_global_load_async_to_lds_b128(gsrc, ldst, 0, 0);
#else
  *(v8h*)dst = *(const v8h*)src;
#endif
}

// ---------------------------------------------------------------------------
// Weight fp32 [K,N] -> f16 transposed [N,K] (makes all B-fragments contiguous)
// ---------------------------------------------------------------------------
__global__ void wtrans_kernel(const float* __restrict__ w, _Float16* __restrict__ wT,
                              int K, int N) {
  int gid = blockIdx.x * blockDim.x + threadIdx.x;
  if (gid >= K * N) return;
  int k = gid / N, n = gid - k * N;
  wT[(size_t)n * K + k] = (_Float16)w[gid];
}

// ---------------------------------------------------------------------------
// Relative-position bias grid: bg[h][i][j] = table[idx(i,j)*HEADS + h]
// idx(i,j) = (yi-yj+15)*31 + (xi-xj+15) for 16x16 grid
// ---------------------------------------------------------------------------
__global__ void biasgrid_kernel(const float* __restrict__ table, float* __restrict__ bg) {
  int gid = blockIdx.x * blockDim.x + threadIdx.x;   // HEADS*NTOK*NTOK
  if (gid >= HEADS * NTOK * NTOK) return;
  int j = gid & 255, i = (gid >> 8) & 255, h = gid >> 16;
  int idx = ((i >> 4) - (j >> 4) + 15) * 31 + ((i & 15) - (j & 15) + 15);
  bg[gid] = table[idx * HEADS + h];
}

// ---------------------------------------------------------------------------
// Fused NCHW->tokens transpose + both LayerNorms (LN1 for attn, LN2 for FFN)
// ---------------------------------------------------------------------------
__global__ __launch_bounds__(256) void ln_kernel(
    const float* __restrict__ x,
    const float* __restrict__ g1, const float* __restrict__ b1,
    const float* __restrict__ g2, const float* __restrict__ b2,
    float* __restrict__ t, _Float16* __restrict__ l1, _Float16* __restrict__ l2) {
  __shared__ float red_s[8], red_q[8];
  int row = blockIdx.x;
  int b = row >> 8, p = row & 255;
  int tid = threadIdx.x;
  int c0 = tid, c1 = tid + 256;
  float v0 = x[((size_t)b * C_SZ + c0) * NTOK + p];
  bool has1 = (c1 < C_SZ);
  float v1 = has1 ? x[((size_t)b * C_SZ + c1) * NTOK + p] : 0.0f;
  float s = v0 + v1, q = v0 * v0 + v1 * v1;
  for (int off = 1; off < 32; off <<= 1) {
    s += __shfl_xor(s, off, 32);
    q += __shfl_xor(q, off, 32);
  }
  int lane = tid & 31, wave = tid >> 5;
  if (lane == 0) { red_s[wave] = s; red_q[wave] = q; }
  __syncthreads();
  s = 0.0f; q = 0.0f;
#pragma unroll
  for (int w = 0; w < 8; ++w) { s += red_s[w]; q += red_q[w]; }
  float mean = s * (1.0f / C_SZ);
  float var  = q * (1.0f / C_SZ) - mean * mean;
  float rstd = rsqrtf(var + 1e-5f);
  size_t rb = (size_t)row * C_SZ;
  t[rb + c0] = v0;
  float n0 = (v0 - mean) * rstd;
  l1[rb + c0] = (_Float16)(n0 * g1[c0] + b1[c0]);
  l2[rb + c0] = (_Float16)(n0 * g2[c0] + b2[c0]);
  if (has1) {
    t[rb + c1] = v1;
    float n1 = (v1 - mean) * rstd;
    l1[rb + c1] = (_Float16)(n1 * g1[c1] + b1[c1]);
    l2[rb + c1] = (_Float16)(n1 * g2[c1] + b2[c1]);
  }
}

// ---------------------------------------------------------------------------
// Register-blocked WMMA GEMM: out[M,N] = A[M,K](f16) @ BT[N,K]^T (+bias)(+GELU)
// Block = 8 waves covering 128(M) x 64(N); each wave owns a 16x64 strip with
// 4 v8f accumulators, reusing one A-fragment per 4 WMMAs. The shared 64x32 B
// tile is double-buffered in LDS via async global->LDS DMA (ASYNCcnt), so the
// next tile's DMA overlaps the current tile's v_wmma issue.
// All grid dims divide exactly; no divergent exits (barriers inside).
// ---------------------------------------------------------------------------
__global__ __launch_bounds__(256) void gemm_wmma_kernel(
    const _Float16* __restrict__ A, const _Float16* __restrict__ BT,
    const float* __restrict__ bias, float* __restrict__ outF,
    _Float16* __restrict__ outH, int M, int N, int K, int gelu) {
  __shared__ _Float16 bstage[2][64 * 32];   // 2 x 4 KB B tiles [col][k]
  int tid  = threadIdx.x;
  int lane = tid & 31;
  int wave = tid >> 5;
  int row0 = blockIdx.x * 128 + wave * 16;
  int col0 = blockIdx.y * 64;

  v8f acc[4] = {{}, {}, {}, {}};
  int nsteps = K >> 5;

  stage_b(BT, bstage[0], K, col0, 0, tid);            // ASYNCcnt = 1
  for (int i = 0; i < nsteps; ++i) {
    bool more = (i + 1 < nsteps);
    if (more) stage_b(BT, bstage[(i + 1) & 1], K, col0, (i + 1) << 5, tid);
    if (more) wait_async_le1();                       // current tile resident
    else      wait_async_le0();
    __syncthreads();                                  // all waves' DMA visible
    v16h a = load_a_frag(A, K, row0, i << 5, lane);
    const _Float16* cur = bstage[i & 1];
#pragma unroll
    for (int t = 0; t < 4; ++t) {
      v16h b = load_b_frag(cur, 32, t * 16, 0, lane);
      acc[t] = __builtin_amdgcn_wmma_f32_16x16x32_f16(false, a, false, b,
                                                      (short)0, acc[t], false, false);
    }
    __syncthreads();                                  // before buffer re-stage
  }

  int colb  = lane & 15;
  int rbase = row0 + ((lane >> 4) << 3);
#pragma unroll
  for (int t = 0; t < 4; ++t) {
    int col = col0 + t * 16 + colb;
    float bv = bias ? bias[col] : 0.0f;
#pragma unroll
    for (int r = 0; r < 8; ++r) {
      float v = acc[t][r] + bv;
      if (gelu) v = 0.5f * v * (1.0f + erff(v * 0.70710678f));   // exact GELU
      size_t oi = (size_t)(rbase + r) * N + col;
      if (outH) outH[oi] = (_Float16)v;
      else      outF[oi] = v;
    }
  }
}

// ---------------------------------------------------------------------------
// Attention: one block per (batch, head). 4 waves, each wave owns 16-query
// row blocks. scores (16x256) live in registers; softmax via 16-lane xor
// shuffles; probs staged f16 in per-wave LDS to re-fragment for P@V WMMA.
// V kept transposed in LDS so B-fragments are contiguous.
// ---------------------------------------------------------------------------
__global__ __launch_bounds__(128) void attn_kernel(
    const _Float16* __restrict__ qkv, const float* __restrict__ biasg,
    _Float16* __restrict__ o) {
  __shared__ _Float16 vT[DH * NTOK];        // 16 KB : vT[d][p]
  __shared__ _Float16 probs[4 * 16 * NTOK]; // 32 KB : per-wave [16][256]

  int lane = threadIdx.x & 31;
  int wave = threadIdx.x >> 5;              // 0..3
  int b = blockIdx.x >> 3, h = blockIdx.x & 7;
  const _Float16* base = qkv + (size_t)b * NTOK * QKV_W;
  const _Float16* qb = base + h * DH;               // q[p][d] at p*768
  const _Float16* kb = base + INNER + h * DH;       // k[p][d] at p*768

  for (int idx = threadIdx.x; idx < NTOK * DH; idx += 128) {
    int p = idx >> 5, d = idx & 31;
    vT[d * NTOK + p] = base[(size_t)p * QKV_W + 2 * INNER + h * DH + d];
  }
  __syncthreads();

  _Float16* pw = probs + wave * 16 * NTOK;
  int colb = lane & 15;
  int rloc = (lane >> 4) << 3;

  for (int rb = wave; rb < 16; rb += 4) {
    int row0 = rb * 16;
    // --- scores: q(16x32) @ k^T, one WMMA per 16-col tile, K=DH=32 ---
    float sc[16][8];
    v16h aq = load_a_frag(qb, QKV_W, row0, 0, lane);
#pragma unroll
    for (int ct = 0; ct < 16; ++ct) {
      v16h bk = load_b_frag(kb, QKV_W, ct * 16, 0, lane);
      v8f d8 = {};
      d8 = __builtin_amdgcn_wmma_f32_16x16x32_f16(false, aq, false, bk, (short)0,
                                                  d8, false, false);
      const float* bp = biasg + ((size_t)h * NTOK + row0 + rloc) * NTOK + ct * 16 + colb;
#pragma unroll
      for (int r = 0; r < 8; ++r)
        sc[ct][r] = d8[r] * 0.17677669529663687f + bp[(size_t)r * NTOK];
    }
    // --- softmax per row: element r shares a row within each 16-lane half,
    //     so xor-shuffle masks 1,2,4,8 reduce the row ---
#pragma unroll
    for (int r = 0; r < 8; ++r) {
      float m = sc[0][r];
#pragma unroll
      for (int ct = 1; ct < 16; ++ct) m = fmaxf(m, sc[ct][r]);
      for (int off = 1; off < 16; off <<= 1) m = fmaxf(m, __shfl_xor(m, off, 32));
      float s = 0.0f;
#pragma unroll
      for (int ct = 0; ct < 16; ++ct) {
        float e = __expf(sc[ct][r] - m);
        sc[ct][r] = e;
        s += e;
      }
      for (int off = 1; off < 16; off <<= 1) s += __shfl_xor(s, off, 32);
      float inv = 1.0f / s;
#pragma unroll
      for (int ct = 0; ct < 16; ++ct) sc[ct][r] *= inv;
    }
    // --- stage probs to LDS (row-major 16x256 f16) for A-fragment reload ---
#pragma unroll
    for (int ct = 0; ct < 16; ++ct)
#pragma unroll
      for (int r = 0; r < 8; ++r)
        pw[(rloc + r) * NTOK + ct * 16 + colb] = (_Float16)sc[ct][r];
    asm volatile("s_wait_dscnt 0" ::: "memory");
    // --- out(16x32) = P(16x256) @ V(256x32): 2 N-tiles x 8 K-steps ---
#pragma unroll
    for (int nt = 0; nt < 2; ++nt) {
      v8f acc = {};
#pragma unroll
      for (int ks = 0; ks < 8; ++ks) {
        v16h ap = load_a_frag(pw, NTOK, 0, ks * 32, lane);
        v16h bv = load_b_frag(vT, NTOK, nt * 16, ks * 32, lane);
        acc = __builtin_amdgcn_wmma_f32_16x16x32_f16(false, ap, false, bv,
                                                     (short)0, acc, false, false);
      }
#pragma unroll
      for (int r = 0; r < 8; ++r)
        o[((size_t)b * NTOK + row0 + rloc + r) * INNER + h * DH + nt * 16 + colb] =
            (_Float16)acc[r];
    }
  }
}

// ---------------------------------------------------------------------------
// y = t + attn_out + ff_out, written back NCHW
// ---------------------------------------------------------------------------
__global__ void combine_kernel(const float* __restrict__ t,
                               const float* __restrict__ attn,
                               const float* __restrict__ ff,
                               float* __restrict__ out) {
  int gid = blockIdx.x * blockDim.x + threadIdx.x;   // B*C*NTOK
  if (gid >= B_SZ * C_SZ * NTOK) return;
  int p = gid & 255;
  int rest = gid >> 8;
  int c = rest % C_SZ;
  int b = rest / C_SZ;
  size_t ri = ((size_t)b * NTOK + p) * C_SZ + c;
  out[gid] = t[ri] + attn[ri] + ff[ri];
}

// ---------------------------------------------------------------------------
extern "C" void kernel_launch(void* const* d_in, const int* in_sizes, int n_in,
                              void* d_out, int out_size, void* d_ws, size_t ws_size,
                              hipStream_t stream) {
  const float* x     = (const float*)d_in[0];
  const float* ln1_g = (const float*)d_in[1];
  const float* ln1_b = (const float*)d_in[2];
  const float* w_qkv = (const float*)d_in[3];
  const float* btab  = (const float*)d_in[4];
  const float* w_out = (const float*)d_in[5];
  const float* b_out = (const float*)d_in[6];
  const float* ln2_g = (const float*)d_in[7];
  const float* ln2_b = (const float*)d_in[8];
  const float* w_ff1 = (const float*)d_in[9];
  const float* b_ff1 = (const float*)d_in[10];
  const float* w_ff2 = (const float*)d_in[11];
  const float* b_ff2 = (const float*)d_in[12];
  float* out = (float*)d_out;
  (void)in_sizes; (void)n_in; (void)out_size; (void)ws_size;

  char* ws = (char*)d_ws;
  size_t off = 0;
  auto alloc = [&](size_t bytes) -> void* {
    void* p = ws + off;
    off += (bytes + 255) & ~(size_t)255;
    return p;
  };
  float*    t_f   = (float*)   alloc((size_t)MTOT * C_SZ * 4);
  float*    att_f = (float*)   alloc((size_t)MTOT * C_SZ * 4);
  float*    ff_f  = (float*)   alloc((size_t)MTOT * C_SZ * 4);
  _Float16* l1    = (_Float16*)alloc((size_t)MTOT * C_SZ * 2);
  _Float16* l2    = (_Float16*)alloc((size_t)MTOT * C_SZ * 2);
  _Float16* qkv_h = (_Float16*)alloc((size_t)MTOT * QKV_W * 2);
  _Float16* o_h   = (_Float16*)alloc((size_t)MTOT * INNER * 2);
  _Float16* h1_h  = (_Float16*)alloc((size_t)MTOT * HID * 2);
  _Float16* wqkvT = (_Float16*)alloc((size_t)C_SZ * QKV_W * 2);
  _Float16* woutT = (_Float16*)alloc((size_t)INNER * C_SZ * 2);
  _Float16* wff1T = (_Float16*)alloc((size_t)C_SZ * HID * 2);
  _Float16* wff2T = (_Float16*)alloc((size_t)HID * C_SZ * 2);
  float*    bg    = (float*)   alloc((size_t)HEADS * NTOK * NTOK * 4);

  // weight conversion/transpose + bias grid
  wtrans_kernel<<<(C_SZ * QKV_W + 255) / 256, 256, 0, stream>>>(w_qkv, wqkvT, C_SZ, QKV_W);
  wtrans_kernel<<<(INNER * C_SZ + 255) / 256, 256, 0, stream>>>(w_out, woutT, INNER, C_SZ);
  wtrans_kernel<<<(C_SZ * HID + 255) / 256, 256, 0, stream>>>(w_ff1, wff1T, C_SZ, HID);
  wtrans_kernel<<<(HID * C_SZ + 255) / 256, 256, 0, stream>>>(w_ff2, wff2T, HID, C_SZ);
  biasgrid_kernel<<<(HEADS * NTOK * NTOK + 255) / 256, 256, 0, stream>>>(btab, bg);

  // transpose + both layernorms
  ln_kernel<<<MTOT, 256, 0, stream>>>(x, ln1_g, ln1_b, ln2_g, ln2_b, t_f, l1, l2);

  // qkv = LN1(t) @ w_qkv          [16384,384] x [384,768] -> f16
  gemm_wmma_kernel<<<dim3(MTOT / 128, QKV_W / 64), 256, 0, stream>>>(
      l1, wqkvT, nullptr, nullptr, qkv_h, MTOT, QKV_W, C_SZ, 0);

  // attention per (batch, head)
  attn_kernel<<<B_SZ * HEADS, 128, 0, stream>>>(qkv_h, bg, o_h);

  // attn_out = o @ w_out + b_out  [16384,256] x [256,384] -> fp32
  gemm_wmma_kernel<<<dim3(MTOT / 128, C_SZ / 64), 256, 0, stream>>>(
      o_h, woutT, b_out, att_f, nullptr, MTOT, C_SZ, INNER, 0);

  // h1 = GELU(LN2(t) @ w_ff1 + b_ff1)   [16384,384] x [384,1536] -> f16
  gemm_wmma_kernel<<<dim3(MTOT / 128, HID / 64), 256, 0, stream>>>(
      l2, wff1T, b_ff1, nullptr, h1_h, MTOT, HID, C_SZ, 1);

  // ff_out = h1 @ w_ff2 + b_ff2   [16384,1536] x [1536,384] -> fp32
  gemm_wmma_kernel<<<dim3(MTOT / 128, C_SZ / 64), 256, 0, stream>>>(
      h1_h, wff2T, b_ff2, ff_f, nullptr, MTOT, C_SZ, HID, 0);

  // y = t + attn_out + ff_out, NCHW writeback
  combine_kernel<<<(B_SZ * C_SZ * NTOK + 255) / 256, 256, 0, stream>>>(
      t_f, att_f, ff_f, out);
}
